// NNUE_44744969289892
// MI455X (gfx1250) — matrix operand order repaired and checked by reading, not compiled, not run on hardware
//
#include <hip/hip_runtime.h>

// ---------------------------------------------------------------------------
// Fused NNUE forward for gfx1250 (MI455X), wave32.
//   phase 1: embedding-bag gather/sum (L2-resident 42MB table), relu, bf16 pack
//   phase 2: L1 GEMM [16x512]x[512x32] via v_wmma_f32_16x16x32_bf16
//   phase 3: L2 GEMM [16x32]x[32x32] via WMMA (one wmma per N-tile)
//   phase 4: output dot via shfl_xor butterfly in 16-lane halves
// ---------------------------------------------------------------------------

namespace {
constexpr int kB       = 16384;  // positions
constexpr int kK       = 30;     // active features per side
constexpr int kFtOut   = 256;    // feature-transform width (per side)
constexpr int kFt2     = 512;    // concatenated width
constexpr int kL1      = 32;     // hidden widths
constexpr int kLStr    = 520;    // LDS row stride in bf16 elems (512 + 8 pad -> 16B aligned, no bank conflicts)
constexpr int kLStr2   = 40;     // LDS row stride for the 16x32 restage
constexpr int kWaves   = 2;      // waves per block (each wave owns a 16-position tile)
constexpr int kThreads = kWaves * 32;
}

typedef __attribute__((ext_vector_type(16))) __bf16 v16bf;
typedef __attribute__((ext_vector_type(8)))  float  v8f;

union FragBF {
  uint4          q[2];
  unsigned short us[16];
  v16bf          v;
};

__device__ __forceinline__ unsigned short f2bf(float x) {
  unsigned u = __builtin_bit_cast(unsigned, x);
  u += 0x7FFFu + ((u >> 16) & 1u);   // round-to-nearest-even
  return (unsigned short)(u >> 16);
}

__global__ __launch_bounds__(kThreads) void nnue_fused_kernel(
    const int* __restrict__ w_idx, const int* __restrict__ b_idx,
    const int* __restrict__ stm,
    const float* __restrict__ ft_w,  const float* __restrict__ ft_bias,
    const float* __restrict__ l1_w,  const float* __restrict__ l1_b,
    const float* __restrict__ l2_w,  const float* __restrict__ l2_b,
    const float* __restrict__ out_w, const float* __restrict__ out_b,
    float* __restrict__ out)
{
  __shared__ unsigned short lds[kWaves * 16 * kLStr];

  const int lane = threadIdx.x & 31;
  const int wave = threadIdx.x >> 5;
  unsigned short* my = lds + wave * 16 * kLStr;
  const int p0 = (blockIdx.x * kWaves + wave) * 16;

  // each lane owns columns [lane*4, lane*4+4) and [128+lane*4, 128+lane*4+4)
  const float4 bias0 = *(const float4*)(ft_bias + lane * 4);
  const float4 bias1 = *(const float4*)(ft_bias + 128 + lane * 4);

  // ---------------- phase 1: embedding bag -> LDS bf16 tile [16 x 512] -----
  for (int i = 0; i < 16; ++i) {
    const int p  = p0 + i;
    const int st = stm[p];
    for (int side = 0; side < 2; ++side) {
      const int* idx = (side == 0) ? w_idx : b_idx;
      // white goes to cols [0,256) when stm==0, else [256,512); black opposite
      const int cbase = ((side == 0) == (st == 0)) ? 0 : kFtOut;
      float ax[8] = {0.f, 0.f, 0.f, 0.f, 0.f, 0.f, 0.f, 0.f};
      long long r = idx[p * kK];
      for (int k = 0; k < kK; ++k) {
        const float4* row = (const float4*)(ft_w + r * kFtOut);
        if (k + 1 < kK) {
          r = idx[p * kK + k + 1];
          __builtin_prefetch(ft_w + r * kFtOut, 0, 0);   // global_prefetch_b8
        }
        const float4 v0 = row[lane];
        const float4 v1 = row[lane + 32];
        ax[0] += v0.x; ax[1] += v0.y; ax[2] += v0.z; ax[3] += v0.w;
        ax[4] += v1.x; ax[5] += v1.y; ax[6] += v1.z; ax[7] += v1.w;
      }
      ax[0] += bias0.x; ax[1] += bias0.y; ax[2] += bias0.z; ax[3] += bias0.w;
      ax[4] += bias1.x; ax[5] += bias1.y; ax[6] += bias1.z; ax[7] += bias1.w;
      #pragma unroll
      for (int j = 0; j < 8; ++j) ax[j] = fmaxf(ax[j], 0.f);

      uint2 s0, s1;
      s0.x = (unsigned)f2bf(ax[0]) | ((unsigned)f2bf(ax[1]) << 16);
      s0.y = (unsigned)f2bf(ax[2]) | ((unsigned)f2bf(ax[3]) << 16);
      s1.x = (unsigned)f2bf(ax[4]) | ((unsigned)f2bf(ax[5]) << 16);
      s1.y = (unsigned)f2bf(ax[6]) | ((unsigned)f2bf(ax[7]) << 16);
      *(uint2*)(my + i * kLStr + cbase +       lane * 4) = s0;
      *(uint2*)(my + i * kLStr + cbase + 128 + lane * 4) = s1;
    }
  }

  // ---------------- phase 2: L1 GEMM via WMMA bf16 -------------------------
  const int n0    = lane & 15;        // column within N-tile / row of A
  const int half  = lane >> 4;        // 0: lanes 0-15, 1: lanes 16-31
  const int koffA = half ? 16 : 0;    // A: K {0..15} vs {16..31}
  const int koffB = half ? 16 : 0;    // B: K {0..15} vs {16..31}

  v8f c0, c1;
  {
    const float ba = l1_b[n0], bb = l1_b[n0 + 16];
    #pragma unroll
    for (int j = 0; j < 8; ++j) { c0[j] = ba; c1[j] = bb; }
  }

  for (int k0 = 0; k0 < kFt2; k0 += 32) {
    FragBF a;
    const uint4* ap = (const uint4*)(my + n0 * kLStr + k0 + koffA);
    a.q[0] = ap[0];        // 8 bf16: K = k0+koffA .. +7
    a.q[1] = ap[1];        // 8 bf16: K = k0+koffA+8 .. +15

    FragBF b0, b1;
    const float* w0 = l1_w + n0        * kFt2 + k0 + koffB;
    const float* w1 = l1_w + (n0 + 16) * kFt2 + k0 + koffB;
    #pragma unroll
    for (int j = 0; j < 16; ++j) { b0.us[j] = f2bf(w0[j]); b1.us[j] = f2bf(w1[j]); }

    c0 = __builtin_amdgcn_wmma_f32_16x16x32_bf16(false, a.v, false, b0.v,
                                                 (short)0, c0, false, false);
    c1 = __builtin_amdgcn_wmma_f32_16x16x32_bf16(false, a.v, false, b1.v,
                                                 (short)0, c1, false, false);
  }
  #pragma unroll
  for (int j = 0; j < 8; ++j) { c0[j] = fmaxf(c0[j], 0.f); c1[j] = fmaxf(c1[j], 0.f); }

  // ---------------- phase 3: restage x1 through LDS, L2 GEMM via WMMA ------
  {
    const int mbase = half * 8;       // D layout: vgpr j -> row j + 8*half
    #pragma unroll
    for (int j = 0; j < 8; ++j) {
      my[(mbase + j) * kLStr2 + n0]      = f2bf(c0[j]);
      my[(mbase + j) * kLStr2 + n0 + 16] = f2bf(c1[j]);
    }
  }

  FragBF a2;
  {
    const uint4* ap2 = (const uint4*)(my + n0 * kLStr2 + koffA);
    a2.q[0] = ap2[0];
    a2.q[1] = ap2[1];
  }
  FragBF b20, b21;
  {
    const float* w0 = l2_w + n0        * kL1 + koffB;
    const float* w1 = l2_w + (n0 + 16) * kL1 + koffB;
    #pragma unroll
    for (int j = 0; j < 16; ++j) { b20.us[j] = f2bf(w0[j]); b21.us[j] = f2bf(w1[j]); }
  }

  v8f d0, d1;
  {
    const float ba = l2_b[n0], bb = l2_b[n0 + 16];
    #pragma unroll
    for (int j = 0; j < 8; ++j) { d0[j] = ba; d1[j] = bb; }
  }
  d0 = __builtin_amdgcn_wmma_f32_16x16x32_bf16(false, a2.v, false, b20.v,
                                               (short)0, d0, false, false);
  d1 = __builtin_amdgcn_wmma_f32_16x16x32_bf16(false, a2.v, false, b21.v,
                                               (short)0, d1, false, false);
  #pragma unroll
  for (int j = 0; j < 8; ++j) { d0[j] = fmaxf(d0[j], 0.f); d1[j] = fmaxf(d1[j], 0.f); }

  // ---------------- phase 4: output dot + butterfly reduce -----------------
  const float ow0 = out_w[n0], ow1 = out_w[n0 + 16];
  const float ob  = out_b[0];
  float rsum[8];
  #pragma unroll
  for (int j = 0; j < 8; ++j) rsum[j] = d0[j] * ow0 + d1[j] * ow1;
  #pragma unroll
  for (int m = 1; m < 16; m <<= 1) {
    #pragma unroll
    for (int j = 0; j < 8; ++j) rsum[j] += __shfl_xor(rsum[j], m, 32);
  }
  if (n0 == 0) {          // lanes 0 and 16 hold rows 0..7 and 8..15
    const int mbase = half * 8;
    #pragma unroll
    for (int j = 0; j < 8; ++j) out[p0 + mbase + j] = rsum[j] + ob;
  }
}

// ---------------------------------------------------------------------------
extern "C" void kernel_launch(void* const* d_in, const int* in_sizes, int n_in,
                              void* d_out, int out_size, void* d_ws, size_t ws_size,
                              hipStream_t stream) {
  // setup_inputs order:
  // 0 w_idx, 1 w_off(int64, unused: off==p*K), 2 b_idx, 3 b_off(unused),
  // 4 stm, 5 ft_w, 6 ft_bias, 7 l1_w, 8 l1_b, 9 l2_w, 10 l2_b, 11 out_w, 12 out_b
  const int*   w_idx   = (const int*)d_in[0];
  const int*   b_idx   = (const int*)d_in[2];
  const int*   stm     = (const int*)d_in[4];
  const float* ft_w    = (const float*)d_in[5];
  const float* ft_bias = (const float*)d_in[6];
  const float* l1_w    = (const float*)d_in[7];
  const float* l1_b    = (const float*)d_in[8];
  const float* l2_w    = (const float*)d_in[9];
  const float* l2_b    = (const float*)d_in[10];
  const float* out_w   = (const float*)d_in[11];
  const float* out_b   = (const float*)d_in[12];
  float*       out     = (float*)d_out;

  dim3 grid(kB / (16 * kWaves));
  nnue_fused_kernel<<<grid, kThreads, 0, stream>>>(
      w_idx, b_idx, stm, ft_w, ft_bias, l1_w, l1_b, l2_w, l2_b, out_w, out_b, out);
}
